// ClassicInteractionBlock_69054484185713
// MI455X (gfx1250) — matrix operand order repaired and checked by reading, not compiled, not run on hardware
//
#include <hip/hip_runtime.h>
#include <math.h>

// Problem constants (match reference)
#define B_   64
#define N_   100
#define M_   99
#define MP   112   // M padded to 7 tiles of 16
#define NIN_ 32
#define NG_  32
#define NF_  64

#define HLD  72    // padded f16 LDS row stride (halves); 144B row = 16B multiple
#define CLD  68    // padded f32 LDS row stride for conv features (272B = 16B mult)

typedef __attribute__((ext_vector_type(16))) _Float16 v16h;
typedef __attribute__((ext_vector_type(8)))  _Float16 v8h;
typedef __attribute__((ext_vector_type(8)))  float    v8f;

union AFrag { v16h v; v8h g[2]; _Float16 h[16]; };
union CFrag { v8f  v; float f[8]; };

// ---- WMMA wrapper: D = A(16x32 f16) * B(32x16 f16) + C(f32) --------------
__device__ __forceinline__ v8f wmma16(v16h a, v16h b, v8f c) {
  return __builtin_amdgcn_wmma_f32_16x16x32_f16(
      /*neg_a=*/false, a, /*neg_b=*/false, b,
      /*c_mod=*/(short)0, c, /*reuse_a=*/false, /*reuse_b=*/false);
}

// A-matrix 16x32 f16 fragment from LDS row-major [16][ld] at K offset koff.
// Lane L: M = L%16 ; halves 0..7 -> K = kb..kb+7 ; halves 8..15 -> K = kb+16..kb+23
// Both groups contiguous -> 2 x ds_load_b128.
__device__ __forceinline__ v16h frag_a_lds(const _Float16* base, int ld,
                                           int lane, int koff) {
  AFrag a;
  int m  = lane & 15;
  int kb = koff + ((lane >> 4) << 3);
  const _Float16* p = base + m * ld + kb;
  a.g[0] = *(const v8h*)(p);
  a.g[1] = *(const v8h*)(p + 16);
  return a.v;
}

// A fragment from a global f32 row (row pointer already includes M): 4 x float4.
__device__ __forceinline__ v16h frag_a_g32(const float* row, int lane, int koff) {
  AFrag a;
  int kb = koff + ((lane >> 4) << 3);
  float4 f0 = *(const float4*)(row + kb);
  float4 f1 = *(const float4*)(row + kb + 4);
  float4 f2 = *(const float4*)(row + kb + 16);
  float4 f3 = *(const float4*)(row + kb + 20);
  a.h[0]=(_Float16)f0.x; a.h[1]=(_Float16)f0.y; a.h[2]=(_Float16)f0.z; a.h[3]=(_Float16)f0.w;
  a.h[4]=(_Float16)f1.x; a.h[5]=(_Float16)f1.y; a.h[6]=(_Float16)f1.z; a.h[7]=(_Float16)f1.w;
  a.h[8]=(_Float16)f2.x; a.h[9]=(_Float16)f2.y; a.h[10]=(_Float16)f2.z; a.h[11]=(_Float16)f2.w;
  a.h[12]=(_Float16)f3.x; a.h[13]=(_Float16)f3.y; a.h[14]=(_Float16)f3.z; a.h[15]=(_Float16)f3.w;
  return a.v;
}

// B fragment from LDS pre-swizzled storage: lane's 16 halves are contiguous.
// frag id = ktile*ntiles + ntile ; one aligned 32B load -> 2 x ds_load_b128.
__device__ __forceinline__ v16h frag_b_sw(const _Float16* wsw, int frag, int lane) {
  return *(const v16h*)(wsw + (((frag << 5) + lane) << 4));
}

// Cooperative fill of swizzled B-fragment storage from a global row-major
// f32 weight W[K][64].  nfrag = (K/32)*4 fragments of 512 halves.
// half j of lane in frag (kt,nt):  K = kt*32 + (lane/16)*8 + j + (j/8)*8,
//                                  N = nt*16 + lane%16
__device__ __forceinline__ void fill_b_sw(_Float16* wsw, const float* W,
                                          int nfrag, int tid, int nthreads) {
  int total = nfrag << 9;
  for (int t = tid; t < total; t += nthreads) {
    int frag = t >> 9;
    int rem  = t & 511;
    int lane = rem >> 4;
    int j    = rem & 15;
    int kt   = frag >> 2;
    int nt   = frag & 3;
    int K = (kt << 5) + ((lane >> 4) << 3) + j + ((j >> 3) << 3);
    int N = (nt << 4) + (lane & 15);
    wsw[t] = (_Float16)W[K * NF_ + N];
  }
}

__device__ __forceinline__ void zero_c(CFrag& c) {
#pragma unroll
  for (int r = 0; r < 8; ++r) c.f[r] = 0.f;
}

// ===========================================================================
// Kernel 1: x = features @ W0   [6400,32] @ [32,64] -> [6400,64]
// 400 row-tiles of 16; one wave per tile; 4 WMMA per wave.
// ===========================================================================
__global__ void __launch_bounds__(256)
k_init_dense(const float* __restrict__ features, const float* __restrict__ W0,
             float* __restrict__ xout) {
  __shared__ alignas(32) _Float16 w0sw[4 * 512];
  fill_b_sw(w0sw, W0, 4, threadIdx.x, blockDim.x);
  __syncthreads();

  int lane = threadIdx.x & 31;
  int wave = threadIdx.x >> 5;
  int row0 = (blockIdx.x * 8 + wave) * 16;  // 6400 rows total, exact tiles

  v16h a = frag_a_g32(features + (size_t)(row0 + (lane & 15)) * NIN_, lane, 0);
#pragma unroll
  for (int nt = 0; nt < 4; ++nt) {
    CFrag c; zero_c(c);
    c.v = wmma16(a, frag_b_sw(w0sw, nt, lane), c.v);
    int n = nt * 16 + (lane & 15);
#pragma unroll
    for (int r = 0; r < 8; ++r) {
      int m = row0 + r + ((lane >> 4) << 3);
      xout[(size_t)m * NF_ + n] = c.f[r];
    }
  }
}

// ===========================================================================
// Kernel 2: fused cfconv + attention per atom.
// 1 block per (b,atom); 7 waves; each wave owns one 16-row neighbor tile.
//   h    = tanh(rbf @ W1 + b1)                (4 WMMA / wave)
//   filt = h @ W2 + b2                        (8 WMMA / wave)
//   conv = x[nbr] * filt ; softmax over M; agg = attn @ conv
// ===========================================================================
__global__ void __launch_bounds__(224)
k_cfconv(const float* __restrict__ rbf, const int* __restrict__ nbr,
         const float* __restrict__ x,
         const float* __restrict__ W1, const float* __restrict__ b1,
         const float* __restrict__ W2, const float* __restrict__ b2,
         const float* __restrict__ nbrf,
         float* __restrict__ attn_out, float* __restrict__ agg_out) {
  __shared__ alignas(32) _Float16 w1sw[4 * 512];
  __shared__ alignas(32) _Float16 w2sw[8 * 512];
  __shared__ alignas(32) _Float16 hstage[7][16 * HLD];
  __shared__ alignas(16) float conv[MP * CLD];
  __shared__ float b1s[NF_], b2s[NF_], nfs[NF_];
  __shared__ float logits[MP];
  __shared__ float attn_s[MP];
  __shared__ float red[2];

  int tid = threadIdx.x;
  int bn  = blockIdx.x;          // b*N + atom
  int b   = bn / N_;

  fill_b_sw(w1sw, W1, 4, tid, blockDim.x);
  fill_b_sw(w2sw, W2, 8, tid, blockDim.x);
  if (tid < NF_) { b1s[tid] = b1[tid]; b2s[tid] = b2[tid]; nfs[tid] = nbrf[tid]; }
  __syncthreads();

  int lane = tid & 31;
  int wave = tid >> 5;           // 0..6
  int m0   = wave * 16;

  // ---- GEMM1: rbf tile (16x32) @ W1 (32x64), tanh, stage as f16 ----------
  AFrag a;
  {
    int m = m0 + (lane & 15);
    bool valid = (m < M_);
    const float* rrow = rbf + ((size_t)bn * M_ + m) * NG_;
    if (valid) {
      __builtin_prefetch(rrow, 0, 0);            // global_prefetch_b8
      a.v = frag_a_g32(rrow, lane, 0);
    } else {
#pragma unroll
      for (int j = 0; j < 16; ++j) a.h[j] = (_Float16)0.f;
    }
  }
  _Float16* hrow = hstage[wave];
#pragma unroll
  for (int nt = 0; nt < 4; ++nt) {
    CFrag c; zero_c(c);
    c.v = wmma16(a.v, frag_b_sw(w1sw, nt, lane), c.v);
    int n = nt * 16 + (lane & 15);
    float bias = b1s[n];
#pragma unroll
    for (int r = 0; r < 8; ++r) {
      int ml = r + ((lane >> 4) << 3);
      hrow[ml * HLD + n] = (_Float16)tanhf(c.f[r] + bias);
    }
  }
  __syncthreads();   // hstage visible + phase ordering

  // ---- GEMM2: h (16x64) @ W2 (64x64) + b2, * gathered neighbor feats -----
  v16h a20 = frag_a_lds(hrow, HLD, lane, 0);
  v16h a21 = frag_a_lds(hrow, HLD, lane, 32);
  const int* nrow = nbr + (size_t)bn * M_;
#pragma unroll
  for (int nt = 0; nt < 4; ++nt) {
    CFrag c; zero_c(c);
    c.v = wmma16(a20, frag_b_sw(w2sw, 0 * 4 + nt, lane), c.v);
    c.v = wmma16(a21, frag_b_sw(w2sw, 1 * 4 + nt, lane), c.v);
    int n = nt * 16 + (lane & 15);
    float bias = b2s[n];
#pragma unroll
    for (int r = 0; r < 8; ++r) {
      int ml = r + ((lane >> 4) << 3);
      int m  = m0 + ml;
      float cv = 0.f;
      if (m < M_) {
        int idx   = nrow[m];
        float nfv = x[((size_t)b * N_ + idx) * NF_ + n];
        cv = nfv * (c.f[r] + bias);
      }
      conv[m * CLD + n] = cv;
    }
  }
  __syncthreads();

  // ---- logits over neighbors ---------------------------------------------
  for (int m = tid; m < MP; m += blockDim.x) {
    float s = 0.f;
    if (m < M_) {
      const float* crow = conv + m * CLD;
      for (int n = 0; n < NF_; ++n) s += crow[n] * nfs[n];
    }
    logits[m] = s;
  }
  __syncthreads();

  if (tid == 0) {
    float mx = -1e30f;
    for (int m = 0; m < M_; ++m) mx = fmaxf(mx, logits[m]);
    float sum = 0.f;
    for (int m = 0; m < M_; ++m) sum += __expf(logits[m] - mx);
    red[0] = mx; red[1] = sum;
  }
  __syncthreads();

  {
    float mx = red[0], inv = 1.f / red[1];
    for (int m = tid; m < M_; m += blockDim.x) {
      float aw = __expf(logits[m] - mx) * inv;
      attn_s[m] = aw;
      attn_out[(size_t)bn * M_ + m] = aw;
    }
  }
  __syncthreads();

  // ---- attention-weighted aggregation ------------------------------------
  for (int n = tid; n < NF_; n += blockDim.x) {
    float s = 0.f;
    for (int m = 0; m < M_; ++m) s += attn_s[m] * conv[m * CLD + n];
    agg_out[(size_t)bn * NF_ + n] = s;
  }
}

// ===========================================================================
// Kernel 3: out = tanh(agg @ W3 + b3) @ W4 + b4    [6400,64] rows
// One wave per 16-row tile; 16 WMMA per wave.
// ===========================================================================
__global__ void __launch_bounds__(256)
k_out_dense(const float* __restrict__ agg,
            const float* __restrict__ W3, const float* __restrict__ b3,
            const float* __restrict__ W4, const float* __restrict__ b4,
            float* __restrict__ out) {
  __shared__ alignas(32) _Float16 w3sw[8 * 512];
  __shared__ alignas(32) _Float16 w4sw[8 * 512];
  __shared__ alignas(32) _Float16 stage[8][16 * HLD];
  __shared__ float b3s[NF_], b4s[NF_];

  int tid = threadIdx.x;
  fill_b_sw(w3sw, W3, 8, tid, blockDim.x);
  fill_b_sw(w4sw, W4, 8, tid, blockDim.x);
  if (tid < NF_) { b3s[tid] = b3[tid]; b4s[tid] = b4[tid]; }
  __syncthreads();

  int lane = tid & 31;
  int wave = tid >> 5;
  int row0 = (blockIdx.x * 8 + wave) * 16;

  const float* arow = agg + (size_t)(row0 + (lane & 15)) * NF_;
  v16h a0 = frag_a_g32(arow, lane, 0);
  v16h a1 = frag_a_g32(arow, lane, 32);

  _Float16* srow = stage[wave];
#pragma unroll
  for (int nt = 0; nt < 4; ++nt) {
    CFrag c; zero_c(c);
    c.v = wmma16(a0, frag_b_sw(w3sw, 0 * 4 + nt, lane), c.v);
    c.v = wmma16(a1, frag_b_sw(w3sw, 1 * 4 + nt, lane), c.v);
    int n = nt * 16 + (lane & 15);
    float bias = b3s[n];
#pragma unroll
    for (int r = 0; r < 8; ++r) {
      int ml = r + ((lane >> 4) << 3);
      srow[ml * HLD + n] = (_Float16)tanhf(c.f[r] + bias);
    }
  }
  __syncthreads();

  v16h s0 = frag_a_lds(srow, HLD, lane, 0);
  v16h s1 = frag_a_lds(srow, HLD, lane, 32);
#pragma unroll
  for (int nt = 0; nt < 4; ++nt) {
    CFrag c; zero_c(c);
    c.v = wmma16(s0, frag_b_sw(w4sw, 0 * 4 + nt, lane), c.v);
    c.v = wmma16(s1, frag_b_sw(w4sw, 1 * 4 + nt, lane), c.v);
    int n = nt * 16 + (lane & 15);
    float bias = b4s[n];
#pragma unroll
    for (int r = 0; r < 8; ++r) {
      int m = row0 + r + ((lane >> 4) << 3);
      out[(size_t)m * NF_ + n] = c.f[r] + bias;
    }
  }
}

// ===========================================================================
extern "C" void kernel_launch(void* const* d_in, const int* in_sizes, int n_in,
                              void* d_out, int out_size, void* d_ws,
                              size_t ws_size, hipStream_t stream) {
  const float* features = (const float*)d_in[0];
  const float* rbf      = (const float*)d_in[1];
  const int*   nbr      = (const int*)  d_in[2];
  const float* W0 = (const float*)d_in[3];
  const float* W1 = (const float*)d_in[4];
  const float* b1 = (const float*)d_in[5];
  const float* W2 = (const float*)d_in[6];
  const float* b2 = (const float*)d_in[7];
  const float* nf = (const float*)d_in[8];
  const float* W3 = (const float*)d_in[9];
  const float* b3 = (const float*)d_in[10];
  const float* W4 = (const float*)d_in[11];
  const float* b4 = (const float*)d_in[12];

  float* out  = (float*)d_out;                       // [B*N*NF]
  float* attn = out + (size_t)B_ * N_ * NF_;         // [B*N*M]

  float* x   = (float*)d_ws;                         // [B*N*NF] f32
  float* agg = x + (size_t)B_ * N_ * NF_;            // [B*N*NF] f32

  // (1) initial dense: 400 tiles / 8 waves per block
  k_init_dense<<<50, 256, 0, stream>>>(features, W0, x);
  // (2) fused cfconv + attention: one block per atom
  k_cfconv<<<B_ * N_, 224, 0, stream>>>(rbf, nbr, x, W1, b1, W2, b2, nf,
                                        attn, agg);
  // (3) output dense
  k_out_dense<<<50, 256, 0, stream>>>(agg, W3, b3, W4, b4, out);
}
